// spatialAttention_53446573031793
// MI455X (gfx1250) — matrix operand order repaired and checked by reading, not compiled, not run on hardware
//
#include <hip/hip_runtime.h>
#include <hip/hip_bf16.h>

// ---------------------------------------------------------------------------
// Types for CDNA5 WMMA (wave32): D(16x16 f32) = A(16x32 bf16) x B(32x16 bf16) + C
// ---------------------------------------------------------------------------
typedef __bf16 bf16_t;
typedef __attribute__((ext_vector_type(16))) bf16_t v16bf;
typedef __attribute__((ext_vector_type(8)))  bf16_t v8bf;
typedef __attribute__((ext_vector_type(8)))  float  v8f;
typedef __attribute__((ext_vector_type(4)))  unsigned int u32x4;
typedef __attribute__((ext_vector_type(4)))  int i32x4;
typedef __attribute__((ext_vector_type(8)))  int i32x8;

#if __has_builtin(__builtin_amdgcn_tensor_load_to_lds) && __has_builtin(__builtin_amdgcn_s_wait_tensorcnt)
#define HAVE_TDM 1
#pragma message("CDNA5: TDM tensor_load_to_lds path ENABLED")
#else
#define HAVE_TDM 0
#pragma message("CDNA5: TDM builtins absent, manual LDS staging fallback")
#endif

// A-matrix fragment (16x32 bf16), contiguous k order (same as load_b_frag):
//   lanes 0-15 : row M = lane,    elems 0-15 = K 0-15
//   lanes 16-31: row M = lane-16, elems 0-15 = K 16-31
__device__ static inline v16bf load_a_frag(const bf16_t* base, int ld) {
    const int lane = threadIdx.x & 31;
    const int r    = lane & 15;
    const int kh   = (lane >> 4) << 4;              // 0 or 16
    const bf16_t* p = base + r * ld + kh;
    v8bf lo = *(const v8bf*)(p);                    // K kh .. kh+7
    v8bf hi = *(const v8bf*)(p + 8);                // K kh+8 .. kh+15
    return __builtin_shufflevector(lo, hi, 0,1,2,3,4,5,6,7,8,9,10,11,12,13,14,15);
}

// B-matrix fragment (32x16 bf16):
//   lanes 0-15 : col N = lane,    elems 0-15 = K 0-15
//   lanes 16-31: col N = lane-16, elems 0-15 = K 16-31
// `base` must be K-contiguous per column with leading dim `ld` (elements).
__device__ static inline v16bf load_b_frag(const bf16_t* base, int ld) {
    const int lane = threadIdx.x & 31;
    const int c    = lane & 15;
    const int kh   = (lane >> 4) << 4;              // 0 or 16
    return *(const v16bf*)(base + c * ld + kh);
}

// ---------------------------------------------------------------------------
// Tensor Data Mover: 2D tile (rows x rowlen bf16, row stride in elems) -> LDS.
// ---------------------------------------------------------------------------
#if HAVE_TDM
__device__ static inline unsigned lds_off(const void* p) {
    return (unsigned)(uintptr_t)p;   // low 32 bits of LDS aperture addr = LDS offset
}
__device__ static inline void tdm_load_2d(unsigned lds_byte_off, const bf16_t* gptr,
                                          unsigned rows, unsigned rowlen,
                                          unsigned stride_elems) {
    unsigned long long ga = (unsigned long long)(uintptr_t)gptr;
    u32x4 g0 = { 1u,                                           // count=1, no gather
                 lds_byte_off,                                 // D#.lds_addr
                 (unsigned)ga,                                 // global_addr[31:0]
                 ((unsigned)(ga >> 32) & 0x01FFFFFFu) | (2u << 30) };  // addr[56:32]|type=2
    i32x8 g1 = { (int)(1u << 16),                              // data_size = 2 bytes
                 (int)(rowlen << 16),                          // tensor_dim0 lo16
                 (int)(rows   << 16),                          // dim0 hi | tensor_dim1 lo16
                 (int)(rowlen << 16),                          // dim1 hi | tile_dim0
                 (int)rows,                                    // tile_dim1 | tile_dim2=0
                 (int)stride_elems,                            // tensor_dim0_stride lo32
                 0, 0 };
    i32x4 z4 = {0, 0, 0, 0};
#if defined(__clang_major__) && (__clang_major__ >= 23)
    i32x8 z8 = {0, 0, 0, 0, 0, 0, 0, 0};
    __builtin_amdgcn_tensor_load_to_lds(g0, g1, z4, z4, z8, 0);
#else
    __builtin_amdgcn_tensor_load_to_lds(g0, g1, z4, z4, 0);
#endif
}
#endif

// Manual fallback staging (also compiles when TDM builtins are absent).
__device__ static inline void stage_tile_manual(bf16_t* dst, const bf16_t* src,
                                                int rows, int rowlen, int ld, int tid) {
    const int cpr = rowlen >> 3;  // 16B chunks per row
    for (int c = tid; c < rows * cpr; c += 256) {
        int r = c / cpr, kc = (c - r * cpr) << 3;
        *(int4*)&dst[r * rowlen + kc] = *(const int4*)&src[(size_t)r * ld + kc];
    }
}

// ---------------------------------------------------------------------------
// f32 -> bf16 conversion (row-major copy) and transposing weight conversion.
// ---------------------------------------------------------------------------
__global__ void cvt_f32_bf16(const float* __restrict__ src, bf16_t* __restrict__ dst, int n) {
    int i = blockIdx.x * blockDim.x + threadIdx.x;
    if (i < n) dst[i] = (bf16_t)src[i];
}

// src: [512][512] (k,n) row-major  ->  dst: [n][k] (K-contiguous per output col)
__global__ void cvt_transpose_f32_bf16(const float* __restrict__ src, bf16_t* __restrict__ dst) {
    int i = blockIdx.x * blockDim.x + threadIdx.x;
    if (i < 512 * 512) {
        int k = i >> 9, n = i & 511;
        dst[(size_t)n * 512 + k] = (bf16_t)src[i];
    }
}

// ---------------------------------------------------------------------------
// Tiled bf16 WMMA GEMM:  C[M x 512] = act(A[M x 512] * W[512 x 512] + bias)
// W passed pre-transposed as Bt[n][k]. Block: 256 threads (8 waves).
// C tile 64x128, BK=64, TDM double-buffered LDS staging.
// ---------------------------------------------------------------------------
template <bool RELU, bool OUT_BF16>
__global__ __launch_bounds__(256)
void gemm_bf16_kernel(const bf16_t* __restrict__ A,    // [M x 512] row-major
                      const bf16_t* __restrict__ Bt,   // [512 x 512], [n][k]
                      const float*  __restrict__ bias, // [512]
                      void* __restrict__ Cout, int M) {
    constexpr int KD = 512, ND = 512, NSTEP = KD / 64;
    __shared__ __align__(32) bf16_t As[2][64 * 64];    // [row][k]
    __shared__ __align__(32) bf16_t Bs[2][128 * 64];   // [col][k]

    const int m0   = blockIdx.x * 64;
    const int n0   = blockIdx.y * 128;
    const int tid  = threadIdx.x;
    const int wave = tid >> 5;
    const int lane = tid & 31;
    const int r0   = (wave & 3) * 16;
    const int cb   = (wave >> 2) * 64;

    v8f acc[4];
#pragma unroll
    for (int t = 0; t < 4; ++t)
#pragma unroll
        for (int j = 0; j < 8; ++j) acc[t][j] = 0.0f;

#if HAVE_TDM
    if (tid < 32) {  // wave 0 issues DMA descriptors
        tdm_load_2d(lds_off(&As[0][0]), A  + (size_t)m0 * KD, 64,  64, KD);
        tdm_load_2d(lds_off(&Bs[0][0]), Bt + (size_t)n0 * KD, 128, 64, KD);
    }
#endif
    for (int step = 0; step < NSTEP; ++step) {
        const int cur = step & 1;
#if HAVE_TDM
        if (tid < 32) {
            if (step + 1 < NSTEP) {  // prefetch next tile pair into other buffer
                tdm_load_2d(lds_off(&As[cur ^ 1][0]),
                            A + (size_t)m0 * KD + (step + 1) * 64, 64, 64, KD);
                tdm_load_2d(lds_off(&Bs[cur ^ 1][0]),
                            Bt + (size_t)n0 * KD + (step + 1) * 64, 128, 64, KD);
                __builtin_amdgcn_s_wait_tensorcnt((short)2);  // current pair done
            } else {
                __builtin_amdgcn_s_wait_tensorcnt((short)0);
            }
        }
        __syncthreads();
#else
        stage_tile_manual(&As[cur][0], A  + (size_t)m0 * KD + step * 64, 64,  64, KD, tid);
        stage_tile_manual(&Bs[cur][0], Bt + (size_t)n0 * KD + step * 64, 128, 64, KD, tid);
        __syncthreads();
#endif
#pragma unroll
        for (int kk = 0; kk < 64; kk += 32) {
            v16bf a = load_a_frag(&As[cur][r0 * 64 + kk], 64);
#pragma unroll
            for (int ct = 0; ct < 4; ++ct) {
                v16bf b = load_b_frag(&Bs[cur][(cb + ct * 16) * 64 + kk], 64);
                acc[ct] = __builtin_amdgcn_wmma_f32_16x16x32_bf16(
                    false, a, false, b, (short)0, acc[ct], false, false);
            }
        }
        __syncthreads();
    }

    // Epilogue: bias + activation. D layout: col=lane&15, row=8*(lane>>4)+j.
    const int colc = lane & 15;
    const int rb   = (lane >> 4) << 3;
#pragma unroll
    for (int ct = 0; ct < 4; ++ct) {
        const int gc = n0 + cb + ct * 16 + colc;
        const float bv = bias[gc];
#pragma unroll
        for (int j = 0; j < 8; ++j) {
            const int gr = m0 + r0 + rb + j;
            float v = acc[ct][j] + bv;
            if (RELU) v = v > 0.0f ? v : 0.0f;
            if (OUT_BF16) ((bf16_t*)Cout)[(size_t)gr * ND + gc] = (bf16_t)v;
            else          ((float*) Cout)[(size_t)gr * ND + gc] = v;
        }
    }
}

// ---------------------------------------------------------------------------
// Spatial attention, one workgroup per (b,t,head). N=512 nodes, head dim 64.
// Dynamic LDS: Ql[512][64] | Kl[512][64] | Vt[64][512] (bf16, 64KB each)
//            | S[32][512] f32 (64KB) | Pbf[32][512] bf16 (32KB)
//            | inv_sum[32] | red[256]      -> ~289KB (< 320KB/WGP)
// ---------------------------------------------------------------------------
__global__ __launch_bounds__(256)
void attn_kernel(const bf16_t* __restrict__ Q, const bf16_t* __restrict__ Km,
                 const bf16_t* __restrict__ Vm, bf16_t* __restrict__ O) {
    constexpr int N = 512, D = 512, HD = 64;
    extern __shared__ char smem[];
    bf16_t* Ql    = (bf16_t*)smem;               // [node][d]
    bf16_t* Kl    = (bf16_t*)(smem + 65536);     // [node][d]
    bf16_t* Vt    = (bf16_t*)(smem + 131072);    // [d][node]
    float*  S     = (float*)(smem + 196608);     // [32][N] raw scores / exp
    bf16_t* Pbf   = (bf16_t*)(smem + 262144);    // [32][N] normalized probs
    float*  inv_s = (float*)(smem + 294912);     // [32]
    float*  red   = (float*)(smem + 295040);     // [256]

    const int head = blockIdx.x & 7;
    const int bt   = blockIdx.x >> 3;
    const size_t base = (size_t)bt * N * D + (size_t)head * HD;
    const int tid  = threadIdx.x;
    const int wave = tid >> 5;
    const int lane = tid & 31;

#if HAVE_TDM
    if (tid < 32) {  // TDM: stage Q and K head-slices row-major into LDS
        tdm_load_2d(lds_off(Ql), Q  + base, N, HD, D);
        tdm_load_2d(lds_off(Kl), Km + base, N, HD, D);
    }
#else
    stage_tile_manual(Ql, Q  + base, N, HD, D, tid);
    stage_tile_manual(Kl, Km + base, N, HD, D, tid);
#endif
    // Stage V transposed [d][node] (B-frag for P*V reads K(=node)-contiguous).
    for (int c = tid; c < (N * HD) / 8; c += 256) {
        int node = c >> 3, dd = (c & 7) << 3;
        int4 t4 = *(const int4*)&Vm[base + (size_t)node * D + dd];
        const bf16_t* tb = (const bf16_t*)&t4;
#pragma unroll
        for (int j = 0; j < 8; ++j) Vt[(dd + j) * N + node] = tb[j];
    }
#if HAVE_TDM
    if (tid < 32) __builtin_amdgcn_s_wait_tensorcnt((short)0);
#endif
    __syncthreads();

    const float scale = 0.125f;  // 1/sqrt(64)
    const int colc = lane & 15;
    const int rb   = (lane >> 4) << 3;

    for (int q0 = 0; q0 < N; q0 += 32) {
        // S[0:32][0:512] = Q_tile x K^T. Per wave: row tile tr = wave&1 fixed,
        // A-frags hoisted; two col tiles in flight for independent WMMA chains.
        {
            const int tr = wave & 1;
            v16bf a0 = load_a_frag(&Ql[(q0 + tr * 16) * HD + 0],  HD);
            v16bf a1 = load_a_frag(&Ql[(q0 + tr * 16) * HD + 32], HD);
            for (int t = wave; t < 64; t += 16) {
                const int tc0 = t >> 1, tc1 = (t + 8) >> 1;
                v8f acc0, acc1;
#pragma unroll
                for (int j = 0; j < 8; ++j) { acc0[j] = 0.0f; acc1[j] = 0.0f; }
                v16bf b00 = load_b_frag(&Kl[(tc0 * 16) * HD + 0],  HD);
                v16bf b01 = load_b_frag(&Kl[(tc0 * 16) * HD + 32], HD);
                v16bf b10 = load_b_frag(&Kl[(tc1 * 16) * HD + 0],  HD);
                v16bf b11 = load_b_frag(&Kl[(tc1 * 16) * HD + 32], HD);
                acc0 = __builtin_amdgcn_wmma_f32_16x16x32_bf16(
                    false, a0, false, b00, (short)0, acc0, false, false);
                acc1 = __builtin_amdgcn_wmma_f32_16x16x32_bf16(
                    false, a0, false, b10, (short)0, acc1, false, false);
                acc0 = __builtin_amdgcn_wmma_f32_16x16x32_bf16(
                    false, a1, false, b01, (short)0, acc0, false, false);
                acc1 = __builtin_amdgcn_wmma_f32_16x16x32_bf16(
                    false, a1, false, b11, (short)0, acc1, false, false);
#pragma unroll
                for (int j = 0; j < 8; ++j) {
                    S[(tr * 16 + rb + j) * N + tc0 * 16 + colc] = acc0[j] * scale;
                    S[(tr * 16 + rb + j) * N + tc1 * 16 + colc] = acc1[j] * scale;
                }
            }
        }
        __syncthreads();

        // Parallel row softmax: 8 threads per row (row = tid>>3, part = tid&7).
        {
            const int srow = tid >> 3, part = tid & 7;
            float* row = &S[srow * N];
            float pm = -1e30f;
            for (int i = part; i < N; i += 8) pm = fmaxf(pm, row[i]);
            red[tid] = pm;
            __syncthreads();
            float m = red[srow * 8];
#pragma unroll
            for (int j = 1; j < 8; ++j) m = fmaxf(m, red[srow * 8 + j]);
            __syncthreads();
            float ps = 0.0f;
            for (int i = part; i < N; i += 8) {
                float e = __expf(row[i] - m);
                row[i] = e;
                ps += e;
            }
            red[tid] = ps;
            __syncthreads();
            if (part == 0) {
                float s = red[srow * 8];
#pragma unroll
                for (int j = 1; j < 8; ++j) s += red[srow * 8 + j];
                inv_s[srow] = 1.0f / s;
            }
            __syncthreads();
        }

        // Normalize + convert: Pbf[row][col] = bf16(exp * inv_sum), vectorized.
        for (int c = tid; c < (32 * N) / 8; c += 256) {
            const int row = c >> 6, off = (c & 63) << 3;
            const float* sp = &S[row * N + off];
            const float is = inv_s[row];
            v8bf pv;
#pragma unroll
            for (int j = 0; j < 8; ++j) pv[j] = (bf16_t)(sp[j] * is);
            *(v8bf*)&Pbf[row * N + off] = pv;
        }
        __syncthreads();

        // O_tile[32][64] = P x V  (2x4 tiles of 16x16, one per wave)
        {
            const int tr = wave & 1, tc = wave >> 1;
            v8f acc;
#pragma unroll
            for (int j = 0; j < 8; ++j) acc[j] = 0.0f;
            for (int kk = 0; kk < N; kk += 32) {
                v16bf a = load_a_frag(&Pbf[(tr * 16) * N + kk], N);
                v16bf b = load_b_frag(&Vt[(tc * 16) * N + kk], N);
                acc = __builtin_amdgcn_wmma_f32_16x16x32_bf16(
                    false, a, false, b, (short)0, acc, false, false);
            }
#pragma unroll
            for (int j = 0; j < 8; ++j)
                O[base + (size_t)(q0 + tr * 16 + rb + j) * D + tc * 16 + colc] =
                    (bf16_t)acc[j];
        }
        // No trailing barrier needed: next iteration's first softmax barrier
        // rendezvous happens before any wave can overwrite S or Pbf.
    }
}

// ---------------------------------------------------------------------------
// Host-side orchestration
// ---------------------------------------------------------------------------
extern "C" void kernel_launch(void* const* d_in, const int* in_sizes, int n_in,
                              void* d_out, int out_size, void* d_ws, size_t ws_size,
                              hipStream_t stream) {
    const float* X  = (const float*)d_in[0];
    // d_in[1] = STE (unused by the reference)
    const float* Wq = (const float*)d_in[2];
    const float* bq = (const float*)d_in[3];
    const float* Wk = (const float*)d_in[4];
    const float* bk = (const float*)d_in[5];
    const float* Wv = (const float*)d_in[6];
    const float* bv = (const float*)d_in[7];
    const float* W1 = (const float*)d_in[8];
    const float* b1 = (const float*)d_in[9];
    const float* W2 = (const float*)d_in[10];
    const float* b2 = (const float*)d_in[11];
    float* out = (float*)d_out;

    constexpr int M  = 8 * 12 * 512;          // 49152 rows
    constexpr int MD = M * 512;               // 25,165,824 elements
    constexpr int WN = 512 * 512;             // weight elements

    char* ws = (char*)d_ws;
    bf16_t* Xbf = (bf16_t*)ws;  ws += (size_t)MD * 2;
    bf16_t* Wqb = (bf16_t*)ws;  ws += (size_t)WN * 2;   // transposed [n][k]
    bf16_t* Wkb = (bf16_t*)ws;  ws += (size_t)WN * 2;
    bf16_t* Wvb = (bf16_t*)ws;  ws += (size_t)WN * 2;
    bf16_t* W1b = (bf16_t*)ws;  ws += (size_t)WN * 2;
    bf16_t* W2b = (bf16_t*)ws;  ws += (size_t)WN * 2;
    bf16_t* qbf = (bf16_t*)ws;  ws += (size_t)MD * 2;
    bf16_t* kbf = (bf16_t*)ws;  ws += (size_t)MD * 2;
    bf16_t* vbf = (bf16_t*)ws;  ws += (size_t)MD * 2;
    bf16_t* attnO = Xbf;   // X no longer needed after projections
    bf16_t* hbf   = qbf;   // Q no longer needed after attention

    // 1) Precision conversion (weights transposed to [n][k] for TDM staging).
    cvt_f32_bf16<<<(MD + 255) / 256, 256, 0, stream>>>(X, Xbf, MD);
    cvt_transpose_f32_bf16<<<(WN + 255) / 256, 256, 0, stream>>>(Wq, Wqb);
    cvt_transpose_f32_bf16<<<(WN + 255) / 256, 256, 0, stream>>>(Wk, Wkb);
    cvt_transpose_f32_bf16<<<(WN + 255) / 256, 256, 0, stream>>>(Wv, Wvb);
    cvt_transpose_f32_bf16<<<(WN + 255) / 256, 256, 0, stream>>>(W1, W1b);
    cvt_transpose_f32_bf16<<<(WN + 255) / 256, 256, 0, stream>>>(W2, W2b);

    // 2) Q/K/V projections (GEMM + bias + ReLU, bf16 out).
    dim3 ggrid(M / 64, 512 / 128);
    gemm_bf16_kernel<true, true><<<ggrid, 256, 0, stream>>>(Xbf, Wqb, bq, qbf, M);
    gemm_bf16_kernel<true, true><<<ggrid, 256, 0, stream>>>(Xbf, Wkb, bk, kbf, M);
    gemm_bf16_kernel<true, true><<<ggrid, 256, 0, stream>>>(Xbf, Wvb, bv, vbf, M);

    // 3) Attention: one WG per (b,t,head).
    const size_t smem_bytes = 295040 + 1024;   // Ql|Kl|Vt|S|Pbf|inv_s|red
    hipFuncSetAttribute(reinterpret_cast<const void*>(attn_kernel),
                        hipFuncAttributeMaxDynamicSharedMemorySize, (int)smem_bytes);
    attn_kernel<<<96 * 8, 256, smem_bytes, stream>>>(qbf, kbf, vbf, attnO);

    // 4) FFN.
    gemm_bf16_kernel<true,  true ><<<ggrid, 256, 0, stream>>>(attnO, W1b, b1, hbf, M);
    gemm_bf16_kernel<false, false><<<ggrid, 256, 0, stream>>>(hbf,   W2b, b2, out, M);
}
